// GFCN_67430986547264
// MI455X (gfx1250) — matrix operand order repaired and checked by reading, not compiled, not run on hardware
//
#include <hip/hip_runtime.h>
#include <math.h>

typedef __attribute__((ext_vector_type(16))) _Float16 v16h;
typedef __attribute__((ext_vector_type(8)))  float    v8f;

#define H0 512
#define W0 1024
#define H1 512
#define W1C 512
#define H2 256

// ELU with hardware transcendental (v_exp_f32).
__device__ __forceinline__ float eluf(float x) {
    return x > 0.f ? x : (__expf(x) - 1.f);
}

__device__ __forceinline__ float sigmoidf_fast(float x) {
    return __builtin_amdgcn_rcpf(1.f + __expf(-x));
}

// deg in {2,3,4} -> 1/deg without the IEEE divide sequence.
__device__ __forceinline__ float inv_deg(int d) {
    float inv = (d == 4) ? 0.25f : 0.33333334f;
    return (d == 2) ? 0.5f : inv;
}

__device__ __forceinline__ v8f wmma_f16(v16h a, v16h b, v8f c) {
    // D(f32,16x16) = A(f16,16x32) * B(f16,32x16) + C
    return __builtin_amdgcn_wmma_f32_16x16x32_f16(
        /*neg_a=*/false, a, /*neg_b=*/false, b,
        /*c_mod=*/(short)0, c, /*reuse_a=*/false, /*reuse_b=*/false);
}

// ---------------------------------------------------------------------------
// Layer 1 (level-0 spline conv, in=1, out=32) + ELU + horizontal max-pool.
// One thread per level-1 node; computes h for both pooled level-0 columns.
// Direction kernel ids at level 0: left=10, right=14, up=2, down=22.
// ---------------------------------------------------------------------------
__global__ __launch_bounds__(256) void conv0_pool(
    const float* __restrict__ X, const float* __restrict__ W1,
    const float* __restrict__ root1, const float* __restrict__ b1,
    float* __restrict__ X1) {
    __shared__ float sw[6 * 32];  // WL, WR, WU, WD, root, bias
    const int tid = threadIdx.x;
    if (tid < 32) {
        sw[tid]       = W1[10 * 32 + tid];
        sw[32 + tid]  = W1[14 * 32 + tid];
        sw[64 + tid]  = W1[2 * 32 + tid];
        sw[96 + tid]  = W1[22 * 32 + tid];
        sw[128 + tid] = root1[tid];
        sw[160 + tid] = b1[tid];
    }
    __syncthreads();

    const int n = blockIdx.x * blockDim.x + tid;
    if (n >= H1 * W1C) return;
    const int r = n >> 9, c = n & (W1C - 1);
    const int ca = 2 * c, cb = ca + 1;
    const int base = r * W0;

    const float xa  = X[base + ca];
    const float xb  = X[base + cb];
    const float xla = (c > 0)       ? X[base + ca - 1] : 0.f;
    const float xrb = (c < W1C - 1) ? X[base + cb + 1] : 0.f;
    const float xua = (r > 0)       ? X[base - W0 + ca] : 0.f;
    const float xub = (r > 0)       ? X[base - W0 + cb] : 0.f;
    const float xda = (r < H0 - 1)  ? X[base + W0 + ca] : 0.f;
    const float xdb = (r < H0 - 1)  ? X[base + W0 + cb] : 0.f;

    const int rb = (r > 0) + (r < H0 - 1);
    const float ia = inv_deg((c > 0) + 1 + rb);
    const float ib = inv_deg(1 + (c < W1C - 1) + rb);

    float* out = X1 + (size_t)n * 32;
#pragma unroll
    for (int o = 0; o < 32; ++o) {
        const float wl = sw[o], wr = sw[32 + o], wu = sw[64 + o], wd = sw[96 + o];
        const float sa = xla * wl + xb * wr + xua * wu + xda * wd;
        const float sb = xa * wl + xrb * wr + xub * wu + xdb * wd;
        const float ha = eluf(sa * ia + xa * sw[128 + o] + sw[160 + o]);
        const float hb = eluf(sb * ib + xb * sw[128 + o] + sw[160 + o]);
        out[o] = fmaxf(ha, hb);
    }
}

// ---------------------------------------------------------------------------
// Level-1 spline conv via WMMA. One wave handles a 16-node row tile.
// Direction kernel ids at level 1: left=10, right=14, up=7, down=17.
// IN,OUT in {32,64}; RS = row shift for reading through the vertical pool map
// (conv2: RS=0 reads x1 directly; conv3: RS=1 reads pooled x2 via r>>1).
// ---------------------------------------------------------------------------
template <int IN, int OUT, int RS>
__global__ __launch_bounds__(256) void conv_l1(
    const float* __restrict__ X, const float* __restrict__ W25,
    const float* __restrict__ Wroot, const float* __restrict__ bias,
    float* __restrict__ Y) {
    constexpr int KC = IN / 32;   // K chunks of 32
    constexpr int NC = OUT / 16;  // N chunks of 16

    const int lane = threadIdx.x & 31;
    const int m = lane & 15;        // A: M index / B,C: N index
    const int half = lane >> 4;

    // B fragments (weights), loaded once per wave, reused across tiles.
    const float* Wt[5] = {W25 + 10 * IN * OUT, W25 + 14 * IN * OUT,
                          W25 + 7 * IN * OUT,  W25 + 17 * IN * OUT, Wroot};
    v16h fb[5][KC][NC];
#pragma unroll
    for (int t = 0; t < 5; ++t)
#pragma unroll
        for (int kc = 0; kc < KC; ++kc)
#pragma unroll
            for (int nc = 0; nc < NC; ++nc) {
                v16h b = {};
                const float* wp =
                    Wt[t] + (kc * 32 + half * 8) * OUT + nc * 16 + m;
#pragma unroll
                for (int v = 0; v < 8; ++v) {
                    b[2 * v]     = (_Float16)wp[(2 * v + ((v >> 2) << 3)) * OUT];
                    b[2 * v + 1] = (_Float16)wp[(2 * v + 1 + ((v >> 2) << 3)) * OUT];
                }
                fb[t][kc][nc] = b;
            }

    // Per-lane bias values (feature = nc*16 + m), loaded once per wave.
    float bv[NC];
#pragma unroll
    for (int nc = 0; nc < NC; ++nc) bv[nc] = bias[nc * 16 + m];

    const int wave  = (blockIdx.x * blockDim.x + threadIdx.x) >> 5;
    const int nwave = (gridDim.x * blockDim.x) >> 5;
    const int ntiles = H1 * (W1C / 16);

    for (int tile = wave; tile < ntiles; tile += nwave) {
        const int r = tile >> 5;
        const int cbase = (tile & 31) << 4;
        const int c = cbase + m;

        const int rr[5] = {r, r, r - 1, r + 1, r};
        const int cc[5] = {c - 1, c + 1, c, c, c};
        const bool ok[5] = {c > 0, c < W1C - 1, r > 0, r < H1 - 1, true};
        const int rdeg = (r > 0) + (r < H1 - 1);

        // Per-tile 1/degree for the 8 output rows this lane-half owns.
        // Wave-uniform fast path: fully interior tile -> deg==4 everywhere.
        float invv[8];
        if (rdeg == 2 && cbase != 0 && cbase != W1C - 16) {
#pragma unroll
            for (int i = 0; i < 8; ++i) invv[i] = 0.25f;
        } else {
#pragma unroll
            for (int i = 0; i < 8; ++i) {
                const int cn = cbase + i + half * 8;
                invv[i] = inv_deg((cn > 0) + (cn < W1C - 1) + rdeg);
            }
        }

        // A fragments: per-lane row m of neighbor feature matrices.
        v16h fa[5][KC];
#pragma unroll
        for (int t = 0; t < 5; ++t)
#pragma unroll
            for (int kc = 0; kc < KC; ++kc) {
                v16h a = {};
                if (ok[t]) {
                    const float* p =
                        X + ((size_t)((rr[t] >> RS) * W1C + cc[t])) * IN +
                        kc * 32;
#pragma unroll
                    for (int v = 0; v < 8; ++v) {
                        const int kb =
                            ((v >> 2) << 4) + half * 8 + ((v & 3) << 1);
                        a[2 * v]     = (_Float16)p[kb];
                        a[2 * v + 1] = (_Float16)p[kb + 1];
                    }
                }
                fa[t][kc] = a;
            }

#pragma unroll
        for (int nc = 0; nc < NC; ++nc) {
            v8f accN = {};  // neighbor sum (scaled by 1/degree)
            v8f accR = {};  // root term
#pragma unroll
            for (int kc = 0; kc < KC; ++kc) {
                accN = wmma_f16(fa[0][kc], fb[0][kc][nc], accN);
                accN = wmma_f16(fa[1][kc], fb[1][kc][nc], accN);
                accN = wmma_f16(fa[2][kc], fb[2][kc][nc], accN);
                accN = wmma_f16(fa[3][kc], fb[3][kc][nc], accN);
                accR = wmma_f16(fa[4][kc], fb[4][kc][nc], accR);
            }
#pragma unroll
            for (int i = 0; i < 8; ++i) {
                const int cn = cbase + i + half * 8;  // node column
                const int f = nc * 16 + m;            // output feature
                const float val = accN[i] * invv[i] + accR[i] + bv[nc];
                Y[((size_t)(r * W1C + cn)) * OUT + f] = eluf(val);
            }
        }
    }
}

// ---------------------------------------------------------------------------
// Vertical 2:1 max-pool of h1 -> x2 (feature-wise).
// ---------------------------------------------------------------------------
__global__ __launch_bounds__(256) void pool2(const float* __restrict__ H1f,
                                             float* __restrict__ X2) {
    const int t = blockIdx.x * blockDim.x + threadIdx.x;
    if (t >= H2 * W1C * 64) return;
    const int f = t & 63;
    const int node2 = t >> 6;
    const int c = node2 & (W1C - 1);
    const int p = node2 >> 9;
    const size_t a = ((size_t)((2 * p) * W1C + c)) * 64 + f;
    X2[t] = fmaxf(H1f[a], H1f[a + (size_t)W1C * 64]);
}

// ---------------------------------------------------------------------------
// Layer 4 (level-0 conv, in=32, out=1) through the horizontal unpool map + sigmoid.
// Level-0 ids: left=10, right=14, up=2, down=22.
// ---------------------------------------------------------------------------
__global__ __launch_bounds__(256) void conv3_out(
    const float* __restrict__ H3, const float* __restrict__ W4,
    const float* __restrict__ root4, const float* __restrict__ b4,
    float* __restrict__ OUT) {
    __shared__ float sw[5 * 32 + 1];
    const int tid = threadIdx.x;
    if (tid < 32) {
        sw[tid]       = W4[10 * 32 + tid];
        sw[32 + tid]  = W4[14 * 32 + tid];
        sw[64 + tid]  = W4[2 * 32 + tid];
        sw[96 + tid]  = W4[22 * 32 + tid];
        sw[128 + tid] = root4[tid];
        if (tid == 0) sw[160] = b4[0];
    }
    __syncthreads();

    const int n = blockIdx.x * blockDim.x + tid;
    if (n >= H0 * W0) return;
    const int r = n >> 10, c0 = n & (W0 - 1);

    auto hrow = [&](int rr, int cc) {
        return H3 + ((size_t)(rr * W1C + (cc >> 1))) * 32;
    };

    float sN = 0.f, sC = 0.f;
    const float* pc = hrow(r, c0);
#pragma unroll
    for (int k = 0; k < 32; ++k) sC += pc[k] * sw[128 + k];

    if (c0 > 0) {
        const float* p = hrow(r, c0 - 1);
#pragma unroll
        for (int k = 0; k < 32; ++k) sN += p[k] * sw[k];
    }
    if (c0 < W0 - 1) {
        const float* p = hrow(r, c0 + 1);
#pragma unroll
        for (int k = 0; k < 32; ++k) sN += p[k] * sw[32 + k];
    }
    if (r > 0) {
        const float* p = hrow(r - 1, c0);
#pragma unroll
        for (int k = 0; k < 32; ++k) sN += p[k] * sw[64 + k];
    }
    if (r < H0 - 1) {
        const float* p = hrow(r + 1, c0);
#pragma unroll
        for (int k = 0; k < 32; ++k) sN += p[k] * sw[96 + k];
    }

    const float inv =
        inv_deg((c0 > 0) + (c0 < W0 - 1) + (r > 0) + (r < H0 - 1));
    const float val = eluf(sN * inv + sC + sw[160]);
    OUT[n] = sigmoidf_fast(val);
}

// ---------------------------------------------------------------------------
extern "C" void kernel_launch(void* const* d_in, const int* in_sizes, int n_in,
                              void* d_out, int out_size, void* d_ws,
                              size_t ws_size, hipStream_t stream) {
    const float* x     = (const float*)d_in[0];
    const float* W1    = (const float*)d_in[8];
    const float* root1 = (const float*)d_in[9];
    const float* b1    = (const float*)d_in[10];
    const float* W2    = (const float*)d_in[11];
    const float* root2 = (const float*)d_in[12];
    const float* b2    = (const float*)d_in[13];
    const float* W3    = (const float*)d_in[14];
    const float* root3 = (const float*)d_in[15];
    const float* b3    = (const float*)d_in[16];
    const float* W4    = (const float*)d_in[17];
    const float* root4 = (const float*)d_in[18];
    const float* b4    = (const float*)d_in[19];

    float* ws = (float*)d_ws;
    const size_t N1 = (size_t)H1 * W1C;   // 262144
    float* x1 = ws;                        // [N1, 32]
    float* h1 = ws + N1 * 32;              // [N1, 64]
    float* x2 = ws;                        // [N1/2, 64] (reuses x1 region)
    float* h3 = ws + N1 * 32;              // [N1, 32]  (reuses h1 region)

    conv0_pool<<<(H1 * W1C) / 256, 256, 0, stream>>>(x, W1, root1, b1, x1);
    conv_l1<32, 64, 0><<<256, 256, 0, stream>>>(x1, W2, root2, b2, h1);
    pool2<<<(H2 * W1C * 64) / 256, 256, 0, stream>>>(h1, x2);
    conv_l1<64, 32, 1><<<256, 256, 0, stream>>>(x2, W3, root3, b3, h3);
    conv3_out<<<(H0 * W0) / 256, 256, 0, stream>>>(h3, W4, root4, b4,
                                                   (float*)d_out);
}
